// Attention_Param_sharing_Kv_sharing_21071109554262
// MI455X (gfx1250) — matrix operand order repaired and checked by reading, not compiled, not run on hardware
//
#include <hip/hip_runtime.h>

typedef __attribute__((ext_vector_type(2)))  float    v2f;
typedef __attribute__((ext_vector_type(4)))  float    v4f;
typedef __attribute__((ext_vector_type(8)))  float    v8f;
typedef __attribute__((ext_vector_type(4)))  _Float16 v4h;
typedef __attribute__((ext_vector_type(16))) _Float16 v16h;

#define BATCH 4
#define CIN   256
#define NTOK  4096
#define KD    16
#define DH    128
#define EPS   1e-5f
#define PSTR  40   // LDS P row stride in halfs (80B: 16B-aligned chunks, conflict-free)
#define MARGIN 10.0f

// ------- qk projection -> qkT[b][n][16] (A/B fragments contiguous) -------
__global__ __launch_bounds__(256) void proj_qk_kernel(
    const float* __restrict__ x, const float* __restrict__ w,
    const float* __restrict__ g, const float* __restrict__ bb,
    const float* __restrict__ mm, const float* __restrict__ vv,
    float* __restrict__ qkT) {
  int idx = blockIdx.x * blockDim.x + threadIdx.x;   // ((b*16)+o)*1024 + n4
  int n4 = idx & 1023;
  int o  = (idx >> 10) & (KD - 1);
  int b  = idx >> 14;
  const float* xb = x + (size_t)b * CIN * NTOK + n4 * 4;
  const float* wr = w + o * CIN;
  v4f acc = (v4f){};
  #pragma unroll 4
  for (int c = 0; c < CIN; ++c) {
    v4f xv = *(const v4f*)(xb + (size_t)c * NTOK);
    float wc = wr[c];
    acc[0] = fmaf(wc, xv[0], acc[0]);
    acc[1] = fmaf(wc, xv[1], acc[1]);
    acc[2] = fmaf(wc, xv[2], acc[2]);
    acc[3] = fmaf(wc, xv[3], acc[3]);
  }
  float s = g[o] * rsqrtf(vv[o] + EPS);
  float t = bb[o] - mm[o] * s;
  #pragma unroll
  for (int k = 0; k < 4; ++k)
    qkT[((size_t)b * NTOK + n4 * 4 + k) * KD + o] = acc[k] * s + t;
}

// ------- v projection -> vhT[b][dh][n] f16 (B fragments contiguous) -------
__global__ __launch_bounds__(256) void proj_v_kernel(
    const float* __restrict__ x, const float* __restrict__ w,
    const float* __restrict__ g, const float* __restrict__ bb,
    const float* __restrict__ mm, const float* __restrict__ vv,
    _Float16* __restrict__ vhT) {
  int idx = blockIdx.x * blockDim.x + threadIdx.x;   // ((b*128)+o)*1024 + n4
  int n4 = idx & 1023;
  int o  = (idx >> 10) & (DH - 1);
  int b  = idx >> 17;
  const float* xb = x + (size_t)b * CIN * NTOK + n4 * 4;
  const float* wr = w + o * CIN;
  v4f acc = (v4f){};
  #pragma unroll 4
  for (int c = 0; c < CIN; ++c) {
    v4f xv = *(const v4f*)(xb + (size_t)c * NTOK);
    float wc = wr[c];
    acc[0] = fmaf(wc, xv[0], acc[0]);
    acc[1] = fmaf(wc, xv[1], acc[1]);
    acc[2] = fmaf(wc, xv[2], acc[2]);
    acc[3] = fmaf(wc, xv[3], acc[3]);
  }
  float s = g[o] * rsqrtf(vv[o] + EPS);
  float t = bb[o] - mm[o] * s;
  v4h hv;
  #pragma unroll
  for (int k = 0; k < 4; ++k) hv[k] = (_Float16)(acc[k] * s + t);
  *(v4h*)(vhT + ((size_t)b * DH + o) * NTOK + n4 * 4) = hv;
}

// ------- flash attention, lazy online softmax: xr[b][n][dh] = relu(softmax(QK)V) -------
__global__ __launch_bounds__(32) void attn_kernel(
    const float* __restrict__ qkT, const _Float16* __restrict__ vhT,
    float* __restrict__ xr) {
  int b    = blockIdx.x >> 9;          // blocks = 4 * 256 * 2
  int rest = blockIdx.x & 511;
  int qt   = rest >> 1;
  int half = rest & 1;                 // which 64 of the 128 dh columns
  int q0 = qt * 16;
  int lane = threadIdx.x;
  int lm = lane & 15;                  // M row (A) / N col (B,C)
  int hi = lane >> 4;
  const float* qkb = qkT + (size_t)b * NTOK * KD;

  // Q as f32 WMMA A-matrix: per-lane contiguous float pairs
  v2f aq[4];
  const v2f* qrow = (const v2f*)(qkb + (size_t)(q0 + lm) * KD);
  #pragma unroll
  for (int j = 0; j < 4; ++j) aq[j] = qrow[2 * j + hi];

  // V row base pointers for this wave's 4 dh chunks (16 contiguous halfs per frag)
  const _Float16* vbase[4];
  #pragma unroll
  for (int d = 0; d < 4; ++d)
    vbase[d] = vhT + ((size_t)b * DH + (half * 4 + d) * 16 + lm) * NTOK + hi * 16;

  __shared__ _Float16 plds[16 * PSTR];
  // contiguous P position for this lane's two staged values (tile0: K=lm, tile1: K=16+lm)
  int pbase = lm;

  v8f acc[4];
  #pragma unroll
  for (int d = 0; d < 4; ++d) acc[d] = (v8f){};
  float mcur[8], lsum[8];
  #pragma unroll
  for (int r = 0; r < 8; ++r) { mcur[r] = -3.0e38f; lsum[r] = 0.f; }

  for (int kt = 0; kt < NTOK / 32; ++kt) {
    int k0 = kt * 32;
    __builtin_prefetch(qkb + (size_t)(k0 + 64 + lm) * KD, 0, 3);
    __builtin_prefetch(vbase[0] + k0 + 64, 0, 3);
    // two 16x16 score tiles via f32 WMMA
    const v2f* kr0 = (const v2f*)(qkb + (size_t)(k0 + lm) * KD);
    const v2f* kr1 = (const v2f*)(qkb + (size_t)(k0 + 16 + lm) * KD);
    v8f sc0 = (v8f){}, sc1 = (v8f){};
    #pragma unroll
    for (int j = 0; j < 4; ++j) {
      sc0 = __builtin_amdgcn_wmma_f32_16x16x4_f32(false, aq[j], false, kr0[2 * j + hi],
                                                  (short)0, sc0, false, false);
      sc1 = __builtin_amdgcn_wmma_f32_16x16x4_f32(false, aq[j], false, kr1[2 * j + hi],
                                                  (short)0, sc1, false, false);
    }
    // lazy max: the serialized cross-lane reduction runs only when some staged
    // exp would exceed the f16 range; corr==1.0 exactly otherwise, so the
    // rescale arithmetic below is unconditional straight-line code (no acc phi).
    float corr[8];
    bool flag = false;
    #pragma unroll
    for (int r = 0; r < 8; ++r)
      flag = flag || (fmaxf(sc0[r], sc1[r]) > mcur[r] + MARGIN);
    if (__any(flag)) {                 // wave-uniform branch, EXEC stays all-ones
      #pragma unroll
      for (int r = 0; r < 8; ++r) {
        float tmax = fmaxf(sc0[r], sc1[r]);
        #pragma unroll
        for (int msk = 1; msk < 16; msk <<= 1)
          tmax = fmaxf(tmax, __shfl_xor(tmax, msk, 16));
        float nm = fmaxf(mcur[r], tmax);
        corr[r] = __expf(mcur[r] - nm);
        mcur[r] = nm;
      }
    } else {
      #pragma unroll
      for (int r = 0; r < 8; ++r) corr[r] = 1.0f;
    }
    // stage P = exp(s - mcur) in f16 (natural K order -> contiguous A fragments)
    #pragma unroll
    for (int r = 0; r < 8; ++r) {
      float p0 = __expf(sc0[r] - mcur[r]);
      float p1 = __expf(sc1[r] - mcur[r]);
      lsum[r] = lsum[r] * corr[r] + (p0 + p1);
      #pragma unroll
      for (int d = 0; d < 4; ++d) acc[d][r] = acc[d][r] * corr[r];
      int row = r + 8 * hi;
      plds[row * PSTR + pbase]     = (_Float16)p0;
      plds[row * PSTR + pbase + 16] = (_Float16)p1;
    }
    __syncthreads();
    // P as f16 A-matrix 16x32: one contiguous 32B chunk per lane
    v16h pa = *(const v16h*)(plds + lm * PSTR + hi * 16);
    __syncthreads();
    // PV: 4 dh chunks, f16 WMMA f32 accumulate; B fragment = 16 contiguous halfs
    #pragma unroll
    for (int d = 0; d < 4; ++d) {
      v16h vb = *(const v16h*)(vbase[d] + k0);
      acc[d] = __builtin_amdgcn_wmma_f32_16x16x32_f16(false, pa, false, vb,
                                                      (short)0, acc[d], false, false);
    }
  }
  // finalize: reduce lsum across the 16-lane half, normalize, ReLU, store
  #pragma unroll
  for (int r = 0; r < 8; ++r) {
    #pragma unroll
    for (int msk = 1; msk < 16; msk <<= 1)
      lsum[r] += __shfl_xor(lsum[r], msk, 16);
    lsum[r] = 1.0f / lsum[r];
  }
  float* xrb = xr + ((size_t)b * NTOK + q0) * DH;
  #pragma unroll
  for (int d = 0; d < 4; ++d) {
    int col = (half * 4 + d) * 16 + lm;
    #pragma unroll
    for (int r = 0; r < 8; ++r) {
      int row = r + 8 * hi;
      float o = acc[d][r] * lsum[r];
      xrb[(size_t)row * DH + col] = o > 0.f ? o : 0.f;
    }
  }
}

// ------- output projection: out[b][co][n], f32 WMMA K=128, BN folded -------
__global__ __launch_bounds__(32) void proj_out_kernel(
    const float* __restrict__ xr, const float* __restrict__ w_p,
    const float* __restrict__ g, const float* __restrict__ bb,
    const float* __restrict__ mm, const float* __restrict__ vv,
    float* __restrict__ out) {
  int ct = blockIdx.x & 15;            // blocks = 4*256*16
  int qt = (blockIdx.x >> 4) & 255;
  int b  = blockIdx.x >> 12;
  int q0 = qt * 16, c0 = ct * 16;
  int lane = threadIdx.x, lm = lane & 15, hi = lane >> 4;
  const v2f* arow = (const v2f*)(xr + ((size_t)b * NTOK + q0 + lm) * DH);
  const v2f* brow = (const v2f*)(w_p + (size_t)(c0 + lm) * DH);
  v8f c = (v8f){};
  #pragma unroll
  for (int j = 0; j < 32; ++j) {
    c = __builtin_amdgcn_wmma_f32_16x16x4_f32(false, arow[2 * j + hi],
                                              false, brow[2 * j + hi],
                                              (short)0, c, false, false);
  }
  int co = c0 + lm;
  float s = g[co] * rsqrtf(vv[co] + EPS);
  float t = bb[co] - mm[co] * s;
  float* ob = out + ((size_t)b * CIN + co) * NTOK + q0;
  #pragma unroll
  for (int r = 0; r < 8; ++r) ob[r + 8 * hi] = c[r] * s + t;
}

extern "C" void kernel_launch(void* const* d_in, const int* in_sizes, int n_in,
                              void* d_out, int out_size, void* d_ws, size_t ws_size,
                              hipStream_t stream) {
  const float* x    = (const float*)d_in[0];
  // d_in[1] = key_v_input_reduction (unused by the reference)
  const float* w_qk = (const float*)d_in[2];
  const float* g_qk = (const float*)d_in[3];
  const float* b_qk = (const float*)d_in[4];
  const float* m_qk = (const float*)d_in[5];
  const float* v_qk = (const float*)d_in[6];
  const float* w_v  = (const float*)d_in[7];
  const float* g_v  = (const float*)d_in[8];
  const float* b_v  = (const float*)d_in[9];
  const float* m_v  = (const float*)d_in[10];
  const float* v_v  = (const float*)d_in[11];
  const float* w_p  = (const float*)d_in[12];
  const float* g_p  = (const float*)d_in[13];
  const float* b_p  = (const float*)d_in[14];
  const float* m_p  = (const float*)d_in[15];
  const float* v_p  = (const float*)d_in[16];
  float* out = (float*)d_out;

  char* ws = (char*)d_ws;
  float*    qkT = (float*)ws;                       // 1 MB: [b][n][16]
  _Float16* vhT = (_Float16*)(ws + (1u << 20));     // 4 MB: [b][dh][n]
  float*    xr  = (float*)(ws + (5u << 20));        // 8 MB: [b][n][128]

  proj_qk_kernel<<<BATCH * KD * (NTOK / 4) / 256, 256, 0, stream>>>(
      x, w_qk, g_qk, b_qk, m_qk, v_qk, qkT);
  proj_v_kernel<<<BATCH * DH * (NTOK / 4) / 256, 256, 0, stream>>>(
      x, w_v, g_v, b_v, m_v, v_v, vhT);
  attn_kernel<<<BATCH * (NTOK / 16) * 2, 32, 0, stream>>>(qkT, vhT, xr);
  proj_out_kernel<<<BATCH * (NTOK / 16) * (CIN / 16), 32, 0, stream>>>(
      xr, w_p, g_p, b_p, m_p, v_p, out);
}
